// Vocoder_549755814264
// MI455X (gfx1250) — matrix operand order repaired and checked by reading, compiled, not run on hardware
//
#include <hip/hip_runtime.h>
#include <hip/hip_bf16.h>
#include <string.h>

typedef __attribute__((ext_vector_type(16))) _Float16 v16h;
typedef __attribute__((ext_vector_type(8)))  _Float16 v8h;
typedef __attribute__((ext_vector_type(8)))  float    v8f;
typedef __attribute__((ext_vector_type(4)))  unsigned int v4u;
typedef __attribute__((ext_vector_type(8)))  int          v8i;
typedef __attribute__((ext_vector_type(4)))  int          v4i;

#define T_AUD   4096
#define NCH     256
#define NMEL    80
#define TMEL    128
#define UPSK    1024
#define UPSS    256
#define NLAYERS 8
#define NFLOWS  12

// padded activation buffer: 128 zero columns each side -> shifted loads never OOB
#define XPAD 128
#define XLD  (T_AUD + 2 * XPAD)   // 4352

#if defined(__has_builtin)
#if __has_builtin(__builtin_amdgcn_tensor_load_to_lds) && __has_builtin(__builtin_amdgcn_s_wait_tensorcnt)
#define HAS_TDM 1
#else
#define HAS_TDM 0
#endif
#else
#define HAS_TDM 0
#endif

// =====================================================================
// WMMA GEMM: C[M,N] = sum_s A_s[M,K_s] * B_s[K_s,:](cols shifted) + bias
// f16 in, f32 accum, wave32, V_WMMA_F32_16X16X32_F16.
// Workgroup: 256 thr = 8 waves; tile 128(M) x 128(N); wave owns 64x32.
// A tiles staged global->LDS by the Tensor Data Mover (TENSORcnt),
// B tiles staged+transposed by VALU (needs [n][k] layout for fragments).
// =====================================================================
struct GemmSeg { const _Float16* A; const _Float16* B; int K; int shift; int ldb; };
struct GemmParams {
  GemmSeg seg[4];
  int nseg;
  int M, N;
  const float* bias0;
  const float* bias1;
  float* C;
};

#define TM 128
#define TN 128
#define TK 32

__global__ __launch_bounds__(256) void wmma_gemm_kernel(GemmParams p) {
  __shared__ __attribute__((aligned(128))) _Float16 LA[2][TM][TK]; // [buf][m][k]
  __shared__ __attribute__((aligned(128))) _Float16 LB[2][TN][TK]; // [buf][n][k] (transposed)

  const int tid  = threadIdx.x;
  const int lane = tid & 31;
  const int wave = tid >> 5;
  const int wm   = wave >> 2;    // 0..1  -> M offset wm*64
  const int wn   = wave & 3;     // 0..3  -> N offset wn*32
  const int m0   = blockIdx.y * TM;
  const int n0   = blockIdx.x * TN;
  const int lmod  = lane & 15;
  const int lhalf = lane >> 4;   // 0|1

  v8f acc[4][2];
#pragma unroll
  for (int a = 0; a < 4; ++a)
#pragma unroll
    for (int b = 0; b < 2; ++b)
#pragma unroll
      for (int r = 0; r < 8; ++r) acc[a][b][r] = 0.0f;

  for (int s = 0; s < p.nseg; ++s) {
    const _Float16* A = p.seg[s].A;
    const _Float16* B = p.seg[s].B;
    const int K     = p.seg[s].K;
    const int shift = p.seg[s].shift;
    const int ldb   = p.seg[s].ldb;
    const int nk    = K / TK;

    auto load_tile = [&](int kc, int buf) {
      const int k0 = kc * TK;
      // ---- A tile: 128 rows x 32 halves (row-major) via TDM ----
#if HAS_TDM
      if (tid < 32) {
        const unsigned long long ga =
            (unsigned long long)(const void*)(A + (size_t)m0 * K + k0);
        const unsigned int lds =
            (unsigned int)(unsigned long long)(void*)&LA[buf][0][0];
        v4u g0;
        g0[0] = 1u;                                   // count=1, user mode
        g0[1] = lds;                                  // lds_addr
        g0[2] = (unsigned int)(ga & 0xffffffffull);   // global_addr[31:0]
        g0[3] = (unsigned int)((ga >> 32) & 0x1ffffffull) | (2u << 30); // [56:32] | type=2
        v8i g1;
        g1[0] = (1 << 16);                                        // data_size=2B
        g1[1] = (int)(((unsigned)K & 0xffffu) << 16);             // tensor_dim0 lo16
        g1[2] = (int)((((unsigned)K >> 16) & 0xffffu) |
                      (((unsigned)p.M & 0xffffu) << 16));         // td0 hi | td1 lo
        g1[3] = (int)((((unsigned)p.M >> 16) & 0xffffu) |
                      ((unsigned)TK << 16));                      // td1 hi | tile_dim0=32
        g1[4] = TM;                                               // tile_dim1=128, tile_dim2=0
        g1[5] = K;                                                // tensor_dim0_stride lo32
        g1[6] = 0;
        g1[7] = 0;
        v4i gz; gz[0] = 0; gz[1] = 0; gz[2] = 0; gz[3] = 0;
        v8i gz8;
#pragma unroll
        for (int z = 0; z < 8; ++z) gz8[z] = 0;
        // 6-arg toolchain variant: (g0, g1, g2, g3, extra, cpol)
        __builtin_amdgcn_tensor_load_to_lds(g0, g1, gz, gz, gz8, 0);
      }
#else
      {
        const int row = tid >> 1;
        const int off = (tid & 1) * 16;
        const _Float16* src = A + (size_t)(m0 + row) * K + k0 + off;
        v8h a0 = *(const v8h*)(src);
        v8h a1 = *(const v8h*)(src + 8);
        *(v8h*)&LA[buf][row][off]     = a0;
        *(v8h*)&LA[buf][row][off + 8] = a1;
      }
#endif
      // ---- B tile: 32 k-rows x 128 cols, stored transposed LB[n][k] ----
      {
        const int kk      = tid >> 3;          // 0..31
        const int nseg16  = (tid & 7) * 16;    // 0..112
        const int colbase = n0 + nseg16 + shift;
        const _Float16* srow = B + (size_t)(k0 + kk) * ldb;
        _Float16 tmp[16];
        if (shift == 0) {  // 16B-aligned fast path
          v8h b0 = *(const v8h*)(srow + colbase);
          v8h b1 = *(const v8h*)(srow + colbase + 8);
#pragma unroll
          for (int j = 0; j < 8; ++j) { tmp[j] = b0[j]; tmp[8 + j] = b1[j]; }
        } else {           // shifted path: buffer is zero-padded, no guards needed
          const _Float16* sp2 = srow + colbase;
#pragma unroll
          for (int j = 0; j < 16; ++j) tmp[j] = sp2[j];
        }
#pragma unroll
        for (int j = 0; j < 16; ++j) LB[buf][nseg16 + j][kk] = tmp[j];
      }
    };

    load_tile(0, 0);
    for (int kc = 0; kc < nk; ++kc) {
#if HAS_TDM
      if (tid < 32) __builtin_amdgcn_s_wait_tensorcnt(0);
#endif
      __syncthreads();
      if (kc + 1 < nk) load_tile(kc + 1, (kc + 1) & 1);
      const int buf = kc & 1;

      // hoist all fragments, then burst 8 WMMAs
      v16h bfr[2];
#pragma unroll
      for (int nf = 0; nf < 2; ++nf) {
        const int n = wn * 32 + nf * 16 + lmod;
        union { v16h v; v8h h[2]; } bu;
        bu.h[0] = *(const v8h*)&LB[buf][n][lhalf * 8];
        bu.h[1] = *(const v8h*)&LB[buf][n][16 + lhalf * 8];
        bfr[nf] = bu.v;
      }
      v16h afr[4];
#pragma unroll
      for (int mf = 0; mf < 4; ++mf) {
        const int m = wm * 64 + mf * 16 + lmod;
        union { v16h v; v8h h[2]; } au;
        au.h[0] = *(const v8h*)&LA[buf][m][lhalf * 8];
        au.h[1] = *(const v8h*)&LA[buf][m][16 + lhalf * 8];
        afr[mf] = au.v;
      }
#pragma unroll
      for (int mf = 0; mf < 4; ++mf)
#pragma unroll
        for (int nf = 0; nf < 2; ++nf)
          acc[mf][nf] = __builtin_amdgcn_wmma_f32_16x16x32_f16(
              false, afr[mf], false, bfr[nf], (short)0, acc[mf][nf], false, false);
    }
    __syncthreads();  // protect buf reuse across segments
  }

  // ---- epilogue: bias + store f32 ----
#pragma unroll
  for (int mf = 0; mf < 4; ++mf) {
#pragma unroll
    for (int nf = 0; nf < 2; ++nf) {
      const int col = n0 + wn * 32 + nf * 16 + lmod;
#pragma unroll
      for (int r = 0; r < 8; ++r) {
        const int row = m0 + wm * 64 + mf * 16 + lhalf * 8 + r;
        float v = acc[mf][nf][r];
        if (p.bias0) v += p.bias0[row];
        if (p.bias1) v += p.bias1[row];
        p.C[(size_t)row * p.N + col] = v;
      }
    }
  }
}

// =====================================================================
// Small support kernels (bandwidth/FLOP-trivial pieces)
// =====================================================================

__global__ void zero_f16_kernel(_Float16* __restrict__ d, int n) {
  const int id = blockIdx.x * blockDim.x + threadIdx.x;
  if (id < n) d[id] = (_Float16)0.0f;
}

// Transposed-conv upsampler fused with the group/reshape:
// sp_h[(m*8+g)][t] = ups_b[m] + sum_ci sum_p spect[ci,p]*ups_w[ci,m, l-p*256],  l=t*8+g
__global__ void upsample_kernel(const float* __restrict__ spect,
                                const float* __restrict__ ups_w,
                                const float* __restrict__ ups_b,
                                _Float16* __restrict__ sp_h) {
  const int id = blockIdx.x * blockDim.x + threadIdx.x;
  if (id >= NMEL * 8 * T_AUD) return;
  const int t    = id & (T_AUD - 1);
  const int chan = id >> 12;
  const int m = chan >> 3, g = chan & 7;
  const int l = t * 8 + g;
  int p_lo = l - (UPSK - 1) + (UPSS - 1);
  p_lo = p_lo > 0 ? (p_lo >> 8) : 0;
  const int p_hi = min(TMEL - 1, l >> 8);
  float acc = ups_b[m];
  for (int p = p_lo; p <= p_hi; ++p) {
    const int q = l - p * UPSS;  // 0..1023
    for (int ci = 0; ci < NMEL; ++ci)
      acc = fmaf(spect[ci * TMEL + p], ups_w[((size_t)ci * NMEL + m) * UPSK + q], acc);
  }
  sp_h[(size_t)chan * T_AUD + t] = (_Float16)acc;
}

__global__ void f32_to_f16_kernel(const float* __restrict__ s, _Float16* __restrict__ d, int n) {
  const int id = blockIdx.x * blockDim.x + threadIdx.x;
  if (id < n) d[id] = (_Float16)s[id];
}

// in_w (512,256,3) -> 3 contiguous f16 matrices dst[j][o][c]
__global__ void inw_convert_kernel(const float* __restrict__ s, _Float16* __restrict__ d) {
  const int id = blockIdx.x * blockDim.x + threadIdx.x;
  if (id >= 512 * 256 * 3) return;
  const int j = id % 3;
  const int oc = id / 3;  // o*256+c
  d[(size_t)j * 512 * 256 + oc] = (_Float16)s[id];
}

// x = start_w(256,nh) @ a0(nh,T) + start_b ; writes f32 + padded f16 mirror
__global__ void start_kernel(const float* __restrict__ sw, const float* __restrict__ sb,
                             const float* __restrict__ a0, float* __restrict__ Xf,
                             _Float16* __restrict__ Xh, int nh) {
  const int id = blockIdx.x * blockDim.x + threadIdx.x;
  if (id >= NCH * T_AUD) return;
  const int c = id >> 12, t = id & (T_AUD - 1);
  float acc = sb[c];
  for (int j = 0; j < nh; ++j) acc = fmaf(sw[c * nh + j], a0[(size_t)j * T_AUD + t], acc);
  Xf[id] = acc;
  Xh[(size_t)c * XLD + XPAD + t] = (_Float16)acc;
}

// act = tanh(Y[:256]) * sigmoid(Y[256:])
__global__ void gate_kernel(const float* __restrict__ Y, _Float16* __restrict__ act) {
  const int id = blockIdx.x * blockDim.x + threadIdx.x;
  if (id >= NCH * T_AUD) return;
  const float y0 = Y[id];
  const float y1 = Y[id + (size_t)NCH * T_AUD];
  act[id] = (_Float16)(tanhf(y0) * (1.0f / (1.0f + expf(-y1))));
}

// residual/skip update
__global__ void update_kernel(const float* __restrict__ RS, float* __restrict__ Xf,
                              _Float16* __restrict__ Xh, float* __restrict__ OUT,
                              int isFirst, int isLast) {
  const int id = blockIdx.x * blockDim.x + threadIdx.x;
  if (id >= NCH * T_AUD) return;
  const int c = id >> 12, t = id & (T_AUD - 1);
  if (!isLast) {
    const float nx = Xf[id] + RS[id];
    Xf[id] = nx;
    Xh[(size_t)c * XLD + XPAD + t] = (_Float16)nx;
    const float o = RS[id + (size_t)NCH * T_AUD];
    OUT[id] = (isFirst ? 0.0f : OUT[id]) + o;
  } else {
    OUT[id] += RS[id];
  }
}

// out2(2nh,T) = end_w(2nh,256) @ OUT + end_b
__global__ void end_kernel(const float* __restrict__ ew, const float* __restrict__ eb,
                           const float* __restrict__ OUT, float* __restrict__ out2, int rows) {
  const int id = blockIdx.x * blockDim.x + threadIdx.x;
  if (id >= rows * T_AUD) return;
  const int m = id >> 12, t = id & (T_AUD - 1);
  float acc = eb[m];
  for (int c = 0; c < NCH; ++c) acc = fmaf(ew[m * NCH + c], OUT[(size_t)c * T_AUD + t], acc);
  out2[(size_t)m * T_AUD + t] = acc;
}

// a1 = (a1 - bb) * exp(-ss), in place on audio rows [nh, 2nh)
__global__ void couple_kernel(const float* __restrict__ out2, float* __restrict__ audio, int nh) {
  const int id = blockIdx.x * blockDim.x + threadIdx.x;
  if (id >= nh * T_AUD) return;
  const int j = id >> 12, t = id & (T_AUD - 1);
  const float bb = out2[(size_t)j * T_AUD + t];
  const float ss = out2[(size_t)(nh + j) * T_AUD + t];
  float* a = audio + (size_t)(nh + j) * T_AUD + t;
  *a = (*a - bb) * expf(-ss);
}

// Gauss-Jordan inverse of n x n (n <= 8), single thread (tiny)
__global__ void invert_kernel(const float* __restrict__ W, float* __restrict__ INV, int n) {
  if (blockIdx.x != 0 || threadIdx.x != 0) return;
  float a[8][8], b[8][8];
  for (int i = 0; i < n; ++i)
    for (int j = 0; j < n; ++j) { a[i][j] = W[i * n + j]; b[i][j] = (i == j) ? 1.0f : 0.0f; }
  for (int col = 0; col < n; ++col) {
    int piv = col;
    float best = fabsf(a[col][col]);
    for (int r = col + 1; r < n; ++r) {
      const float v = fabsf(a[r][col]);
      if (v > best) { best = v; piv = r; }
    }
    if (piv != col)
      for (int j = 0; j < n; ++j) {
        float t1 = a[col][j]; a[col][j] = a[piv][j]; a[piv][j] = t1;
        float t2 = b[col][j]; b[col][j] = b[piv][j]; b[piv][j] = t2;
      }
    const float ip = 1.0f / a[col][col];
    for (int j = 0; j < n; ++j) { a[col][j] *= ip; b[col][j] *= ip; }
    for (int r = 0; r < n; ++r) {
      if (r == col) continue;
      const float f = a[r][col];
      for (int j = 0; j < n; ++j) { a[r][j] -= f * a[col][j]; b[r][j] -= f * b[col][j]; }
    }
  }
  for (int i = 0; i < n; ++i)
    for (int j = 0; j < n; ++j) INV[i * n + j] = b[i][j];
}

// dst[roff+o][t] = sum_c inv[o,c] * src[c][t]
__global__ void apply_inv_kernel(const float* __restrict__ inv, const float* __restrict__ src,
                                 float* __restrict__ dst, int nc, int roff) {
  const int id = blockIdx.x * blockDim.x + threadIdx.x;
  if (id >= nc * T_AUD) return;
  const int o = id >> 12, t = id & (T_AUD - 1);
  float acc = 0.0f;
  for (int c = 0; c < nc; ++c) acc = fmaf(inv[o * nc + c], src[(size_t)c * T_AUD + t], acc);
  dst[(size_t)(roff + o) * T_AUD + t] = acc;
}

// out[t*8+c] = audio[c][t]
__global__ void interleave_kernel(const float* __restrict__ audio, float* __restrict__ out) {
  const int id = blockIdx.x * blockDim.x + threadIdx.x;
  if (id >= 8 * T_AUD) return;
  const int t = id >> 3, c = id & 7;
  out[id] = audio[(size_t)c * T_AUD + t];
}

// =====================================================================
// Host orchestration
// =====================================================================
static inline int ceil_div(int a, int b) { return (a + b - 1) / b; }

extern "C" void kernel_launch(void* const* d_in, const int* in_sizes, int n_in,
                              void* d_out, int out_size, void* d_ws, size_t ws_size,
                              hipStream_t stream) {
  (void)in_sizes; (void)n_in; (void)out_size; (void)ws_size;
  static const int N_HALF[NFLOWS] = {4, 4, 4, 4, 3, 3, 3, 3, 2, 2, 2, 2};

  const float* spect = (const float*)d_in[0];
  const float* ups_w = (const float*)d_in[1];
  const float* ups_b = (const float*)d_in[2];
  // per-flow params: 38 tensors each, starting at index 3
  auto FP = [&](int k, int item) -> const float* { return (const float*)d_in[3 + k * 38 + item]; };
  // items: 0 start_w 1 start_b 2 cond_w 3 cond_b 4..11 in_w 12..19 in_b
  //        20..27 rs_w 28..35 rs_b 36 end_w 37 end_b
  const int CONV_BASE = 3 + NFLOWS * 38;          // 459
  const float* z_init   = (const float*)d_in[CONV_BASE + NFLOWS];       // 471
  const float* z_early0 = (const float*)d_in[CONV_BASE + NFLOWS + 1];   // 472
  const float* z_early1 = (const float*)d_in[CONV_BASE + NFLOWS + 2];   // 473

  // ---- workspace carve-up ----
  size_t off = 0;
  auto carve = [&](size_t bytes) -> void* {
    void* p = (char*)d_ws + off;
    off += (bytes + 255) & ~(size_t)255;
    return p;
  };
  _Float16* sp_h    = (_Float16*)carve((size_t)640 * T_AUD * 2);
  _Float16* condw_h = (_Float16*)carve((size_t)4096 * 640 * 2);
  _Float16* inw_h   = (_Float16*)carve((size_t)NLAYERS * 3 * 512 * 256 * 2);
  _Float16* rsw_h   = (_Float16*)carve((size_t)NLAYERS * 512 * 256 * 2);
  float*    Ybuf    = (float*)carve((size_t)512 * T_AUD * 4);
  _Float16* act_h   = (_Float16*)carve((size_t)256 * T_AUD * 2);
  float*    RSbuf   = (float*)carve((size_t)512 * T_AUD * 4);
  float*    Xf      = (float*)carve((size_t)256 * T_AUD * 4);
  _Float16* Xh      = (_Float16*)carve((size_t)256 * XLD * 2);   // padded
  float*    OUTbuf  = (float*)carve((size_t)256 * T_AUD * 4);
  float*    out2    = (float*)carve((size_t)8 * T_AUD * 4);
  float*    audioA  = (float*)carve((size_t)8 * T_AUD * 4);
  float*    audioB  = (float*)carve((size_t)8 * T_AUD * 4);
  float*    invm    = (float*)carve((size_t)64 * 4);

  const int B256 = 256;

  // zero padded activation mirror once (interior rewritten by start/update)
  zero_f16_kernel<<<ceil_div(NCH * XLD, B256), B256, 0, stream>>>(Xh, NCH * XLD);

  // upsample + group -> sp (f16)
  upsample_kernel<<<ceil_div(640 * T_AUD, B256), B256, 0, stream>>>(spect, ups_w, ups_b, sp_h);

  // audio <- z_init (4 x T)
  (void)hipMemcpyAsync(audioA, z_init, (size_t)4 * T_AUD * 4, hipMemcpyDeviceToDevice, stream);

  float* cur = audioA;
  float* oth = audioB;

  for (int k = NFLOWS - 1; k >= 0; --k) {
    const int nh = N_HALF[k];

    // ---- convert this flow's weights to f16 ----
    f32_to_f16_kernel<<<ceil_div(4096 * 640, B256), B256, 0, stream>>>(FP(k, 2), condw_h, 4096 * 640);
    for (int i = 0; i < NLAYERS; ++i) {
      inw_convert_kernel<<<ceil_div(512 * 256 * 3, B256), B256, 0, stream>>>(
          FP(k, 4 + i), inw_h + (size_t)i * 3 * 512 * 256);
      const int rs_elems = (i < NLAYERS - 1 ? 512 : 256) * 256;
      f32_to_f16_kernel<<<ceil_div(rs_elems, B256), B256, 0, stream>>>(
          FP(k, 20 + i), rsw_h + (size_t)i * 512 * 256, rs_elems);
    }

    // ---- start 1x1 conv (tiny K) ----
    start_kernel<<<ceil_div(NCH * T_AUD, B256), B256, 0, stream>>>(
        FP(k, 0), FP(k, 1), cur, Xf, Xh, nh);

    // ---- WN layers ----
    for (int i = 0; i < NLAYERS; ++i) {
      const int d = 1 << i;

      // fused: Y = cond_w[512i:+512] @ sp + sum_j Wj @ x(shift) + cond_b_slice + in_b
      GemmParams pa;
      pa.seg[0] = { condw_h + (size_t)512 * i * 640, sp_h, 640, 0, T_AUD };
      pa.seg[1] = { inw_h + (size_t)(i * 3 + 0) * 512 * 256, Xh + XPAD, 256, -d, XLD };
      pa.seg[2] = { inw_h + (size_t)(i * 3 + 1) * 512 * 256, Xh + XPAD, 256, 0, XLD };
      pa.seg[3] = { inw_h + (size_t)(i * 3 + 2) * 512 * 256, Xh + XPAD, 256, +d, XLD };
      pa.nseg = 4;
      pa.M = 512; pa.N = T_AUD;
      pa.bias0 = FP(k, 3) + 512 * i;   // cond_b slice
      pa.bias1 = FP(k, 12 + i);        // in_b
      pa.C = Ybuf;
      wmma_gemm_kernel<<<dim3(T_AUD / TN, 512 / TM), 256, 0, stream>>>(pa);

      gate_kernel<<<ceil_div(NCH * T_AUD, B256), B256, 0, stream>>>(Ybuf, act_h);

      // RS = rs_w @ act + rs_b
      GemmParams pc;
      pc.seg[0] = { rsw_h + (size_t)i * 512 * 256, act_h, 256, 0, T_AUD };
      pc.nseg = 1;
      pc.M = (i < NLAYERS - 1) ? 512 : 256;
      pc.N = T_AUD;
      pc.bias0 = FP(k, 28 + i);
      pc.bias1 = nullptr;
      pc.C = RSbuf;
      wmma_gemm_kernel<<<dim3(T_AUD / TN, pc.M / TM), 256, 0, stream>>>(pc);

      update_kernel<<<ceil_div(NCH * T_AUD, B256), B256, 0, stream>>>(
          RSbuf, Xf, Xh, OUTbuf, (i == 0) ? 1 : 0, (i == NLAYERS - 1) ? 1 : 0);
    }

    // ---- end 1x1 conv (tiny M) ----
    end_kernel<<<ceil_div(2 * nh * T_AUD, B256), B256, 0, stream>>>(
        FP(k, 36), FP(k, 37), OUTbuf, out2, 2 * nh);

    // ---- affine coupling ----
    couple_kernel<<<ceil_div(nh * T_AUD, B256), B256, 0, stream>>>(out2, cur, nh);

    // ---- inverse 1x1 convolution ----
    const float* cw = (const float*)d_in[CONV_BASE + k];
    invert_kernel<<<1, 32, 0, stream>>>(cw, invm, 2 * nh);
    const int concat = (k % 4 == 0 && k > 0) ? 1 : 0;
    apply_inv_kernel<<<ceil_div(2 * nh * T_AUD, B256), B256, 0, stream>>>(
        invm, cur, oth, 2 * nh, concat ? 2 : 0);
    if (concat) {
      const float* ze = (k == 8) ? z_early0 : z_early1;
      (void)hipMemcpyAsync(oth, ze, (size_t)2 * T_AUD * 4, hipMemcpyDeviceToDevice, stream);
    }
    float* t = cur; cur = oth; oth = t;
  }

  interleave_kernel<<<ceil_div(8 * T_AUD, B256), B256, 0, stream>>>(cur, (float*)d_out);
}
